// Head_53558242181246
// MI455X (gfx1250) — matrix operand order repaired and checked by reading, not compiled, not run on hardware
//
#include <hip/hip_runtime.h>
#include <hip/hip_bf16.h>

#define B_ 4
#define T_ 4096
#define C_ 1024
#define H_ 64

typedef __attribute__((ext_vector_type(16))) _Float16 v16h;
typedef __attribute__((ext_vector_type(8)))  _Float16 v8h;
typedef __attribute__((ext_vector_type(8)))  float    v8f;

typedef unsigned int u32x4 __attribute__((ext_vector_type(4)));
typedef int          i32x8 __attribute__((ext_vector_type(8)));
typedef int          i32x4 __attribute__((ext_vector_type(4)));

#if defined(__has_builtin)
#if __has_builtin(__builtin_amdgcn_tensor_load_to_lds) && __has_builtin(__builtin_amdgcn_s_wait_tensorcnt)
#define USE_TDM 1
#endif
#endif
#ifndef USE_TDM
#define USE_TDM 0
#endif

static __device__ __forceinline__ v8f wmma_f16(v16h a, v16h b, v8f c) {
    return __builtin_amdgcn_wmma_f32_16x16x32_f16(
        false, a, false, b, (short)0, c, false, false);
}

// Build a v16h operand from two contiguous 8-half (16B-aligned) chunks.
static __device__ __forceinline__ v16h pack2(const _Float16* lo, const _Float16* hi) {
    v8h a = *(const v8h*)lo;
    v8h b = *(const v8h*)hi;
    return __builtin_shufflevector(a, b, 0,1,2,3,4,5,6,7, 8,9,10,11,12,13,14,15);
}

#if USE_TDM
// TDM: async-load a 64x64 f16 tile (row stride = stride_elems in memory) into
// LDS with a 16B pad after every 128B row -> lands in a [64][72] f16 image.
// D# per CDNA5 ISA (2D tensor, groups 2/3 zero).
// This toolchain exposes the 6-arg builtin: (g0, g1, g2, g3, g4, cpol).
static __device__ __forceinline__ void tdm_load_tile_64x64(
        const _Float16* src, const _Float16* ldsDst, unsigned stride_elems) {
    unsigned long long ga = (unsigned long long)(size_t)src;
    u32x4 g0;
    g0[0] = 1u;                                    // count=1, user descriptor
    g0[1] = (unsigned)(size_t)(const void*)ldsDst; // lds_addr (bytes)
    g0[2] = (unsigned)ga;                          // global_addr[31:0]
    g0[3] = (unsigned)((ga >> 32) & 0x1FFFFFFu) | (2u << 30); // addr[56:32] | type=2
    i32x8 g1;
    // data_size=1 (2B) | pad_enable | pad_interval=4 (128B) | pad_amount=3 (16B)
    g1[0] = (int)((1u << 16) | (1u << 20) | (4u << 22) | (3u << 25));
    g1[1] = (int)(64u << 16);        // tensor_dim0[15:0] = 64 (bits 79:48)
    g1[2] = (int)(64u << 16);        // tensor_dim0 hi = 0 | tensor_dim1 lo = 64
    g1[3] = (int)(64u << 16);        // tensor_dim1 hi = 0 | tile_dim0 = 64
    g1[4] = 64;                      // tile_dim1 = 64, tile_dim2 = 0
    g1[5] = (int)stride_elems;       // tensor_dim0_stride[31:0]
    g1[6] = 0;                       // stride0 hi | tensor_dim1_stride lo
    g1[7] = 0;
    i32x4 z4 = {0, 0, 0, 0};
    i32x8 z8 = {0, 0, 0, 0, 0, 0, 0, 0};
    __builtin_amdgcn_tensor_load_to_lds(g0, g1, z4, z4, z8, 0);
}
#endif

// ---------------------------------------------------------------------------
// Kernel 1: W (f32 [C][H]) -> WMMA B-matrix lane layout, f16.
// wB[m][kk][nt][lane][j]: value = W_m[kk*32+(lane>>4)*16+j][nt*16+(lane&15)]
// ---------------------------------------------------------------------------
__global__ void pack_w(const float* __restrict__ Wq, const float* __restrict__ Wk,
                       const float* __restrict__ Wv, _Float16* __restrict__ wB) {
    int tid = blockIdx.x * blockDim.x + threadIdx.x;
    if (tid >= 3 * 32 * 4 * 32) return;
    int lane = tid & 31;
    int nt   = (tid >> 5) & 3;
    int kk   = (tid >> 7) & 31;
    int m    = tid >> 12;
    const float* W = (m == 0) ? Wq : (m == 1) ? Wk : Wv;
    int n  = nt * 16 + (lane & 15);
    int k0 = kk * 32 + (lane >> 4) * 16;
    v16h o;
#pragma unroll
    for (int j = 0; j < 16; ++j) o[j] = (_Float16)W[(k0 + j) * H_ + n];
    *(v16h*)(wB + (size_t)tid * 16) = o;
}

// ---------------------------------------------------------------------------
// Kernel 2: qkv projection. One wave / 16 rows; 12 accumulators; all 12 B
// tiles loaded into registers first so loads clause up and overlap the WMMAs.
// ---------------------------------------------------------------------------
__global__ void __launch_bounds__(32) qkv_proj(
        const float* __restrict__ x, const _Float16* __restrict__ wB,
        _Float16* __restrict__ q16, _Float16* __restrict__ k16,
        _Float16* __restrict__ vt16) {
    const int lane  = threadIdx.x;
    const int laneN = lane & 15;
    const int half  = lane >> 4;
    const int row0  = blockIdx.x * 16;

    v8f acc[12] = {};
    const float* xrow = x + (size_t)(row0 + laneN) * C_;

    for (int kk = 0; kk < C_ / 32; ++kk) {
        const float* ap = xrow + kk * 32 + half * 16;
        __builtin_prefetch(ap + 256, 0, 1);           // x stream, ~8 iters ahead
        v16h a;
#pragma unroll
        for (int j = 0; j < 8; ++j) {
            a[j]     = (_Float16)ap[j];
            a[j + 8] = (_Float16)ap[j + 8];
        }
        v16h bt[12];
#pragma unroll
        for (int u = 0; u < 12; ++u) {
            int m = u >> 2, nt = u & 3;
            bt[u] = *(const v16h*)(wB + ((((size_t)m * 32 + kk) * 4 + nt) * 32 + lane) * 16);
        }
#pragma unroll
        for (int u = 0; u < 12; ++u) acc[u] = wmma_f16(a, bt[u], acc[u]);
    }

    const int b  = row0 / T_;
    const int t0 = row0 % T_;
#pragma unroll
    for (int nt = 0; nt < 4; ++nt) {
        const int n = nt * 16 + laneN;
#pragma unroll
        for (int i = 0; i < 8; ++i) {
            int r = row0 + i + 8 * half;
            q16[(size_t)r * H_ + n] = (_Float16)acc[0 * 4 + nt][i];
            k16[(size_t)r * H_ + n] = (_Float16)acc[1 * 4 + nt][i];
            int t = t0 + i + 8 * half;
            vt16[((size_t)b * H_ + n) * T_ + t] = (_Float16)acc[2 * 4 + nt][i];
        }
    }
}

// ---------------------------------------------------------------------------
// Kernel 3: causal flash attention. 4 waves / block cover 64 query rows and
// share double-buffered K/V tiles in LDS, streamed in by the Tensor Data
// Mover (s_wait_tensorcnt + barrier producer/consumer).
// ---------------------------------------------------------------------------
__global__ void __launch_bounds__(128) attn(
        const _Float16* __restrict__ q16, const _Float16* __restrict__ k16,
        const _Float16* __restrict__ vt16, float* __restrict__ out) {
    __shared__ _Float16 Kl[2][64][72];   // [s'][h], padded rows
    __shared__ _Float16 Vl[2][64][72];   // [h][s'], padded rows
    __shared__ _Float16 Pls[4][16][72];  // per-wave P staging

    const int tid   = threadIdx.x;
    const int wave  = tid >> 5;
    const int lane  = tid & 31;
    const int laneN = lane & 15;
    const int half  = lane >> 4;
    const int b     = blockIdx.y;
    const int q0    = blockIdx.x * 64 + wave * 16;
    const size_t bT = (size_t)b * T_;
    const _Float16* vbase = vt16 + bT * H_;   // [H][T] for this batch

    // Q as two A-tiles, registers for the whole kernel.
    v16h qa[2];
    {
        const _Float16* qp = q16 + (bT + q0 + laneN) * H_;
#pragma unroll
        for (int kk = 0; kk < 2; ++kk) {
            const _Float16* p = qp + kk * 32 + half * 16;
            qa[kk] = pack2(p, p + 8);
        }
    }

    v8f   o[4] = {};
    float m_i[8], l_i[8];
#pragma unroll
    for (int i = 0; i < 8; ++i) { m_i[i] = -3.0e38f; l_i[i] = 0.0f; }

    const int ntiles = blockIdx.x + 1;   // causal: keys 0 .. blockIdx.x*64+63

#if USE_TDM
    if (wave == 0) {                     // prologue: DMA tile 0
        tdm_load_tile_64x64(k16 + bT * H_, &Kl[0][0][0], H_);
        tdm_load_tile_64x64(vbase,         &Vl[0][0][0], T_);
    }
#endif

    for (int kt = 0; kt < ntiles; ++kt) {
        const int s0 = kt * 64;
#if USE_TDM
        const int buf = kt & 1;
        __builtin_amdgcn_s_wait_tensorcnt(0);   // no-op except issuing wave
        __syncthreads();                        // tile kt visible to all waves
        if (wave == 0 && (kt + 1) < ntiles) {   // DMA tile kt+1 into other buffer
            const int s1 = s0 + 64;
            tdm_load_tile_64x64(k16 + (bT + s1) * H_, &Kl[buf ^ 1][0][0], H_);
            tdm_load_tile_64x64(vbase + s1,           &Vl[buf ^ 1][0][0], T_);
        }
#else
        const int buf = 0;
        __syncthreads();
        // cooperative fallback copy: 128 threads, 2 threads per row
        {
            const int r = tid >> 1, c0 = (tid & 1) * 32;
            const _Float16* ks = k16 + (bT + s0 + r) * H_ + c0;
            const _Float16* vs = vbase + (size_t)r * T_ + s0 + c0;
#pragma unroll
            for (int j = 0; j < 4; ++j) {
                *(v8h*)&Kl[0][r][c0 + j * 8] = *(const v8h*)(ks + j * 8);
                *(v8h*)&Vl[0][r][c0 + j * 8] = *(const v8h*)(vs + j * 8);
            }
        }
        __syncthreads();
#endif

        // ---- S = Q @ K^T ----
        v8f s[4] = {};
#pragma unroll
        for (int kk = 0; kk < 2; ++kk) {
#pragma unroll
            for (int nt = 0; nt < 4; ++nt) {
                const _Float16* kp = &Kl[buf][nt * 16 + laneN][kk * 32 + half * 16];
                s[nt] = wmma_f16(qa[kk], pack2(kp, kp + 8), s[nt]);
            }
        }

        // ---- scale + causal mask ----
        const bool need_mask = (s0 + 63) > q0;
#pragma unroll
        for (int nt = 0; nt < 4; ++nt) {
#pragma unroll
            for (int i = 0; i < 8; ++i) {
                float v = s[nt][i] * 0.125f;   // 1/sqrt(64)
                if (need_mask) {
                    int scol = s0 + nt * 16 + laneN;
                    int qrow = q0 + i + 8 * half;
                    if (scol > qrow) v = -3.0e38f;
                }
                s[nt][i] = v;
            }
        }

        // ---- online softmax (xor<=8 stays inside the 16-lane row groups) ----
#pragma unroll
        for (int i = 0; i < 8; ++i) {
            float v = fmaxf(fmaxf(s[0][i], s[1][i]), fmaxf(s[2][i], s[3][i]));
            v = fmaxf(v, __shfl_xor(v, 1));
            v = fmaxf(v, __shfl_xor(v, 2));
            v = fmaxf(v, __shfl_xor(v, 4));
            v = fmaxf(v, __shfl_xor(v, 8));
            float mn   = fmaxf(m_i[i], v);
            float corr = __expf(m_i[i] - mn);
            m_i[i] = mn;
            float rs = 0.0f;
#pragma unroll
            for (int nt = 0; nt < 4; ++nt) {
                float p = __expf(s[nt][i] - mn);
                s[nt][i] = p;
                rs += p;
            }
            rs += __shfl_xor(rs, 1);
            rs += __shfl_xor(rs, 2);
            rs += __shfl_xor(rs, 4);
            rs += __shfl_xor(rs, 8);
            l_i[i] = l_i[i] * corr + rs;
#pragma unroll
            for (int nt = 0; nt < 4; ++nt) o[nt][i] *= corr;
        }

        // ---- P (C-layout) -> LDS row-major ----
#pragma unroll
        for (int nt = 0; nt < 4; ++nt)
#pragma unroll
            for (int i = 0; i < 8; ++i)
                Pls[wave][i + 8 * half][nt * 16 + laneN] = (_Float16)s[nt][i];
        __syncthreads();

        // ---- O += P @ V ----
#pragma unroll
        for (int kk = 0; kk < 2; ++kk) {
            const _Float16* pp = &Pls[wave][laneN][kk * 32 + half * 16];
            v16h pa = pack2(pp, pp + 8);
#pragma unroll
            for (int nt = 0; nt < 4; ++nt) {
                const _Float16* vp = &Vl[buf][nt * 16 + laneN][kk * 32 + half * 16];
                o[nt] = wmma_f16(pa, pack2(vp, vp + 8), o[nt]);
            }
        }
    }

    // ---- normalize and store f32 [B][T][H] ----
#pragma unroll
    for (int i = 0; i < 8; ++i) {
        float inv = 1.0f / l_i[i];
        int t = q0 + i + 8 * half;
        float* op = out + (bT + t) * H_ + laneN;
#pragma unroll
        for (int nt = 0; nt < 4; ++nt) op[nt * 16] = o[nt][i] * inv;
    }
}

// ---------------------------------------------------------------------------
extern "C" void kernel_launch(void* const* d_in, const int* in_sizes, int n_in,
                              void* d_out, int out_size, void* d_ws, size_t ws_size,
                              hipStream_t stream) {
    (void)in_sizes; (void)n_in; (void)out_size; (void)ws_size;
    const float* x  = (const float*)d_in[0];
    const float* Wq = (const float*)d_in[1];
    const float* Wk = (const float*)d_in[2];
    const float* Wv = (const float*)d_in[3];
    float* out = (float*)d_out;

    char* ws = (char*)d_ws;
    _Float16* wB   = (_Float16*)(ws);                       // 384 KB
    _Float16* q16  = (_Float16*)(ws + 393216);              // 2 MB
    _Float16* k16  = (_Float16*)(ws + 393216 + 2097152);    // 2 MB
    _Float16* vt16 = (_Float16*)(ws + 393216 + 2 * 2097152);// 2 MB, [B][H][T]

    pack_w<<<48, 256, 0, stream>>>(Wq, Wk, Wv, wB);
    qkv_proj<<<(B_ * T_) / 16, 32, 0, stream>>>(x, wB, q16, k16, vt16);
    attn<<<dim3(T_ / 64, B_), 128, 0, stream>>>(q16, k16, vt16, out);
}
